// Cucafera_68685116998338
// MI455X (gfx1250) — compile-verified, matrix operand order checked by reading, not run on hardware
//
#include <hip/hip_runtime.h>
#include <hip/hip_bf16.h>
#include <math.h>

typedef __attribute__((ext_vector_type(16))) __bf16 v16bf;
typedef __attribute__((ext_vector_type(8)))  float  v8f;
typedef unsigned int v4u __attribute__((ext_vector_type(4)));
typedef unsigned int v8u __attribute__((ext_vector_type(8)));

#define T_SEQ  2048
#define DMODEL 768
#define NHEAD  12
#define NKV    4
#define HEADD  64
#define FFDIM  2048
#define NLAYER 4
#define NVOCAB 32000
#define NREP   3

// ---------------------------------------------------------------- WMMA utils

static __device__ __forceinline__ v8f wmma_bf16(v16bf a, v16bf b, v8f c) {
  // D = A(16x32 bf16) * B(32x16 bf16) + C(16x16 f32)
  return __builtin_amdgcn_wmma_f32_16x16x32_bf16(false, a, false, b,
                                                 (short)0, c, false, false);
}

// Fragment loader for K-contiguous bf16 LDS tiles ([row][k], row = m or n).
// CDNA5 16-bit A/B layout: lane<16 -> K = ks+{0..7, 16..23},
// lane>=16 -> K = ks+{8..15, 24..31}. Two contiguous 16B groups per lane.
static __device__ __forceinline__ v16bf lds_frag(const __bf16* rowbase, int ks, int khi) {
  const __bf16* p0 = rowbase + ks + khi * 8;
  v16bf f;
#pragma unroll
  for (int e = 0; e < 8; ++e) { f[e] = p0[e]; f[e + 8] = p0[e + 16]; }
  return f;
}

// Same K mapping but reading fp32 (global or LDS) and converting to bf16.
static __device__ __forceinline__ v16bf frag_from_f32(const float* rowbase, int ks, int khi) {
  const float* p0 = rowbase + ks + khi * 8;
  v16bf f;
#pragma unroll
  for (int e = 0; e < 8; ++e) { f[e] = (__bf16)p0[e]; f[e + 8] = (__bf16)p0[e + 16]; }
  return f;
}

// ---------------------------------------------------------------- GEMM (WMMA)
// C[M,N] = A[M,K] * B[K,N]  (+ optional elementwise Radd). B is K x N row-major.
// M%128 == N%128 == K%64 == 0 for every use in this model.

#define BM 128
#define BN 128
#define BK 64
#define KPAD 72   // bf16 K stride in LDS: 144B rows, 16B aligned

template <int EPI>
__global__ __launch_bounds__(256)
void gemm_bf16_wmma(const float* __restrict__ A, const float* __restrict__ B,
                    const float* __restrict__ Radd, float* __restrict__ C,
                    int M, int N, int K, int lda, int ldb, int ldc) {
  __shared__ __align__(16) __bf16 As[BM][KPAD];   // [m][k]
  __shared__ __align__(16) __bf16 Bs[BN][KPAD];   // [n][k]  (K-contiguous)

  const int tid  = threadIdx.x;
  const int lane = tid & 31;
  const int wave = tid >> 5;          // 0..7
  const int ln   = lane & 15;
  const int khi  = lane >> 4;
  const int waveM = (wave >> 1) * 32; // 4 waves along M
  const int waveN = (wave & 1) * 64;  // 2 waves along N
  const int m0 = blockIdx.y * BM;
  const int n0 = blockIdx.x * BN;

  const v8f z8 = {0.f,0.f,0.f,0.f,0.f,0.f,0.f,0.f};
  v8f acc[2][4];
#pragma unroll
  for (int i = 0; i < 2; ++i)
#pragma unroll
    for (int j = 0; j < 4; ++j) acc[i][j] = z8;

  for (int k0 = 0; k0 < K; k0 += BK) {
    // ---- stage A tile: 128x64 fp32 -> bf16, [m][k]
#pragma unroll
    for (int i = 0; i < 8; ++i) {
      int vv  = tid + i * 256;      // 0..2047 float4 slots
      int row = vv >> 4;            // 16 float4 per row
      int k4  = vv & 15;
      const float4 f4 = *reinterpret_cast<const float4*>(
          A + (size_t)(m0 + row) * lda + k0 + k4 * 4);
      __bf16* dst = &As[row][k4 * 4];
      dst[0] = (__bf16)f4.x; dst[1] = (__bf16)f4.y;
      dst[2] = (__bf16)f4.z; dst[3] = (__bf16)f4.w;
    }
    // ---- stage B tile: vector-load along N, scatter-transpose to [n][k]
#pragma unroll
    for (int i = 0; i < 8; ++i) {
      int vv = tid + i * 256;
      int kk = vv >> 5;             // 32 float4 per k-row (128 n)
      int n4 = vv & 31;
      const float4 f4 = *reinterpret_cast<const float4*>(
          B + (size_t)(k0 + kk) * ldb + n0 + n4 * 4);
      Bs[n4 * 4 + 0][kk] = (__bf16)f4.x;
      Bs[n4 * 4 + 1][kk] = (__bf16)f4.y;
      Bs[n4 * 4 + 2][kk] = (__bf16)f4.z;
      Bs[n4 * 4 + 3][kk] = (__bf16)f4.w;
    }
    if (k0 + BK < K)
      __builtin_prefetch(A + (size_t)(m0 + (tid & 127)) * lda + k0 + BK, 0, 1);
    __syncthreads();

#pragma unroll
    for (int ks = 0; ks < BK; ks += 32) {
      v16bf af[2], bfm[4];
#pragma unroll
      for (int mt = 0; mt < 2; ++mt)
        af[mt] = lds_frag(&As[waveM + mt * 16 + ln][0], ks, khi);
#pragma unroll
      for (int nt = 0; nt < 4; ++nt)
        bfm[nt] = lds_frag(&Bs[waveN + nt * 16 + ln][0], ks, khi);
#pragma unroll
      for (int mt = 0; mt < 2; ++mt)
#pragma unroll
        for (int nt = 0; nt < 4; ++nt)
          acc[mt][nt] = wmma_bf16(af[mt], bfm[nt], acc[mt][nt]);
    }
    __syncthreads();
  }

  // ---- epilogue: C layout row = vgpr + 8*khi, col = lane&15
#pragma unroll
  for (int mt = 0; mt < 2; ++mt) {
#pragma unroll
    for (int nt = 0; nt < 4; ++nt) {
      const int rbase = m0 + waveM + mt * 16 + khi * 8;
      const int c     = n0 + waveN + nt * 16 + ln;
#pragma unroll
      for (int r = 0; r < 8; ++r) {
        size_t o = (size_t)(rbase + r) * ldc + c;
        float val = acc[mt][nt][r];
        if (EPI == 1) val += Radd[o];
        C[o] = val;
      }
    }
  }
}

// --------------------------------------------- TDM-staged B^T GEMM (logits)
// C[M,N] = A[M,K] * B^T where B is N x K row-major (embedding table).
// Both operands are K-contiguous 2D tiles -> stream them HBM->LDS with the
// Tensor Data Mover (TENSORcnt), keeping the vector pipe free for WMMA.

#define TBK  32
#define TKP  36   // fp32 K stride in LDS: 144B rows (TDM pads 4 DW every 32 DW)

// Issue a 2D TDM load: tile (tile_d0 x tile_d1) fp32 elements from gptr
// (row stride = stride_elems) into LDS at byte offset ldsoff.
static __device__ __forceinline__ void tdm_load_2d(const float* gptr, unsigned ldsoff,
                                                   unsigned tensor_d0, unsigned tensor_d1,
                                                   unsigned tile_d0, unsigned tile_d1,
                                                   unsigned long long stride_elems) {
  const unsigned long long ga = (unsigned long long)(size_t)gptr;
  v4u g0;
  g0[0] = 1u;                                          // count=1 (valid user D#)
  g0[1] = ldsoff;                                      // lds_addr (bytes)
  g0[2] = (unsigned)(ga & 0xffffffffu);                // global_addr[31:0]
  g0[3] = (unsigned)((ga >> 32) & 0x1ffffffu)          // global_addr[56:32]
        | (2u << 30);                                  // type = 2 ("image")
  v8u g1;
  g1[0] = (2u << 16)                                   // data_size = 4B
        | (1u << 20)                                   // pad_enable
        | (4u << 22)                                   // pad_interval: 32 DWORDs
        | (3u << 25);                                  // pad_amount: 4 DWORDs
  g1[1] = (tensor_d0 & 0xffffu) << 16;                 // [63:48] = td0 lo16
  g1[2] = (tensor_d0 >> 16) | ((tensor_d1 & 0xffffu) << 16);
  g1[3] = (tensor_d1 >> 16) | (tile_d0 << 16);         // [127:112] = tile_dim0
  g1[4] = tile_d1 & 0xffffu;                           // tile_dim1, tile_dim2=0
  g1[5] = (unsigned)(stride_elems & 0xffffffffu);      // tensor_dim0_stride lo
  g1[6] = (unsigned)((stride_elems >> 32) & 0xffffu);  // stride hi, dim1_stride=0
  g1[7] = 0u;
  asm volatile("tensor_load_to_lds %0, %1" :: "s"(g0), "s"(g1) : "memory");
}

__global__ __launch_bounds__(256)
void gemm_bt_tdm(const float* __restrict__ A, const float* __restrict__ B,
                 float* __restrict__ C, int M, int N, int K,
                 int lda, int ldb, int ldc) {
  __shared__ __align__(16) float As[BM][TKP];   // [m][k] fp32 (TDM-written)
  __shared__ __align__(16) float Bs[BN][TKP];   // [n][k] fp32 (TDM-written)

  const int tid  = threadIdx.x;
  const int lane = tid & 31;
  const int wave = tid >> 5;
  const int ln   = lane & 15;
  const int khi  = lane >> 4;
  const int waveM = (wave >> 1) * 32;
  const int waveN = (wave & 1) * 64;
  const int m0 = blockIdx.y * BM;
  const int n0 = blockIdx.x * BN;

  const v8f z8 = {0.f,0.f,0.f,0.f,0.f,0.f,0.f,0.f};
  v8f acc[2][4];
#pragma unroll
  for (int i = 0; i < 2; ++i)
#pragma unroll
    for (int j = 0; j < 4; ++j) acc[i][j] = z8;

  const unsigned aoff = (unsigned)(size_t)&As[0][0];
  const unsigned boff = (unsigned)(size_t)&Bs[0][0];

  for (int k0 = 0; k0 < K; k0 += TBK) {
    if (wave == 0) {
      tdm_load_2d(A + (size_t)m0 * lda + k0, aoff,
                  (unsigned)(K - k0), (unsigned)BM, TBK, BM,
                  (unsigned long long)lda);
      tdm_load_2d(B + (size_t)n0 * ldb + k0, boff,
                  (unsigned)(K - k0), (unsigned)BN, TBK, BN,
                  (unsigned long long)ldb);
      __builtin_amdgcn_s_wait_tensorcnt(0);
    }
    __syncthreads();

    v16bf af[2], bfm[4];
#pragma unroll
    for (int mt = 0; mt < 2; ++mt)
      af[mt] = frag_from_f32(&As[waveM + mt * 16 + ln][0], 0, khi);
#pragma unroll
    for (int nt = 0; nt < 4; ++nt)
      bfm[nt] = frag_from_f32(&Bs[waveN + nt * 16 + ln][0], 0, khi);
#pragma unroll
    for (int mt = 0; mt < 2; ++mt)
#pragma unroll
      for (int nt = 0; nt < 4; ++nt)
        acc[mt][nt] = wmma_bf16(af[mt], bfm[nt], acc[mt][nt]);
    __syncthreads();
  }

#pragma unroll
  for (int mt = 0; mt < 2; ++mt) {
#pragma unroll
    for (int nt = 0; nt < 4; ++nt) {
      const int rbase = m0 + waveM + mt * 16 + khi * 8;
      const int c     = n0 + waveN + nt * 16 + ln;
#pragma unroll
      for (int r = 0; r < 8; ++r)
        C[(size_t)(rbase + r) * ldc + c] = acc[mt][nt][r];
    }
  }
}

// ---------------------------------------------------------- flash attention
// grid = (T/64, H), block = 128 (4 waves). Each wave: 16 query rows.
// Online-softmax over 32-key tiles; Q*K^T and P*V both on bf16 WMMA.

__global__ __launch_bounds__(128)
void attn_flash(const float* __restrict__ q, const float* __restrict__ k,
                const float* __restrict__ v, float* __restrict__ out) {
  const int h    = blockIdx.y;
  const int kvh  = h / NREP;
  const int q0   = blockIdx.x * 64;
  const int tid  = threadIdx.x;
  const int lane = tid & 31;
  const int wave = tid >> 5;
  const int ln   = lane & 15;
  const int khi  = lane >> 4;
  const int qrow0 = q0 + wave * 16;

  __shared__ __align__(16) __bf16 Ks[32][HEADD + 8];  // [key][dim]
  __shared__ __align__(16) __bf16 Vs[HEADD][32 + 8];  // [dim][key] (V^T)
  __shared__ __align__(16) __bf16 Pb[4][16][40];      // per-wave P transpose buf

  const float* qrow = q + (size_t)(qrow0 + ln) * (NHEAD * HEADD) + h * HEADD;
  const v16bf aq0 = frag_from_f32(qrow, 0, khi);
  const v16bf aq1 = frag_from_f32(qrow, 32, khi);

  const v8f z8 = {0.f,0.f,0.f,0.f,0.f,0.f,0.f,0.f};
  v8f o[4] = {z8, z8, z8, z8};
  float mst[8], lst[8];
#pragma unroll
  for (int i = 0; i < 8; ++i) { mst[i] = -1e30f; lst[i] = 0.f; }

  const int nkb = (q0 + 64) >> 5;   // causal: keys 0 .. q0+63
  for (int kb = 0; kb < nkb; ++kb) {
    const int kbase = kb * 32;
    __syncthreads();
#pragma unroll
    for (int i = 0; i < 4; ++i) {
      int vv  = tid + i * 128;      // 0..511 float4 slots (32 keys x 16 f4)
      int key = vv >> 4;
      int d4  = vv & 15;
      const size_t src = (size_t)(kbase + key) * (NKV * HEADD) + (size_t)kvh * HEADD + d4 * 4;
      const float4 fk = *reinterpret_cast<const float4*>(k + src);
      __bf16* dk = &Ks[key][d4 * 4];
      dk[0] = (__bf16)fk.x; dk[1] = (__bf16)fk.y;
      dk[2] = (__bf16)fk.z; dk[3] = (__bf16)fk.w;
      const float4 fv = *reinterpret_cast<const float4*>(v + src);
      Vs[d4 * 4 + 0][key] = (__bf16)fv.x;
      Vs[d4 * 4 + 1][key] = (__bf16)fv.y;
      Vs[d4 * 4 + 2][key] = (__bf16)fv.z;
      Vs[d4 * 4 + 3][key] = (__bf16)fv.w;
    }
    __syncthreads();

    v8f s0 = z8, s1 = z8;
    s0 = wmma_bf16(aq0, lds_frag(&Ks[ln][0],      0,  khi), s0);
    s0 = wmma_bf16(aq1, lds_frag(&Ks[ln][0],      32, khi), s0);
    s1 = wmma_bf16(aq0, lds_frag(&Ks[16 + ln][0], 0,  khi), s1);
    s1 = wmma_bf16(aq1, lds_frag(&Ks[16 + ln][0], 32, khi), s1);

    const int ki0 = kbase + ln;
    const int ki1 = kbase + 16 + ln;
#pragma unroll
    for (int r = 0; r < 8; ++r) {
      const int qi = qrow0 + r + 8 * khi;
      float a = s0[r] * 0.125f; if (ki0 > qi) a = -1e30f;   // causal mask
      float b = s1[r] * 0.125f; if (ki1 > qi) b = -1e30f;
      float rm = fmaxf(a, b);
      rm = fmaxf(rm, __shfl_xor(rm, 1, 32));
      rm = fmaxf(rm, __shfl_xor(rm, 2, 32));
      rm = fmaxf(rm, __shfl_xor(rm, 4, 32));
      rm = fmaxf(rm, __shfl_xor(rm, 8, 32));
      const float mnew = fmaxf(mst[r], rm);
      const float sc   = __expf(mst[r] - mnew);
      const float pa   = __expf(a - mnew);
      const float pb   = __expf(b - mnew);
      float rs = pa + pb;
      rs += __shfl_xor(rs, 1, 32);
      rs += __shfl_xor(rs, 2, 32);
      rs += __shfl_xor(rs, 4, 32);
      rs += __shfl_xor(rs, 8, 32);
      lst[r] = lst[r] * sc + rs;
      mst[r] = mnew;
#pragma unroll
      for (int nt = 0; nt < 4; ++nt) o[nt][r] *= sc;
      Pb[wave][r + 8 * khi][ln]      = (__bf16)pa;
      Pb[wave][r + 8 * khi][16 + ln] = (__bf16)pb;
    }
    // intra-wave cross-lane LDS dependency: explicit CDNA5 dscnt wait
    asm volatile("s_wait_dscnt 0" ::: "memory");
    const v16bf ap = lds_frag(&Pb[wave][ln][0], 0, khi);
#pragma unroll
    for (int nt = 0; nt < 4; ++nt)
      o[nt] = wmma_bf16(ap, lds_frag(&Vs[nt * 16 + ln][0], 0, khi), o[nt]);
  }

#pragma unroll
  for (int nt = 0; nt < 4; ++nt) {
#pragma unroll
    for (int r = 0; r < 8; ++r) {
      const int row = qrow0 + r + 8 * khi;
      out[(size_t)row * (NHEAD * HEADD) + h * HEADD + nt * 16 + ln] =
          o[nt][r] / lst[r];
    }
  }
}

// ---------------------------------------------------------------- small ops

__global__ void embed_kernel(const int* __restrict__ idx, const float* __restrict__ emb,
                             float* __restrict__ x, int n) {
  for (int i = blockIdx.x * blockDim.x + threadIdx.x; i < n;
       i += gridDim.x * blockDim.x) {
    const int t = i / DMODEL, d = i - t * DMODEL;
    x[i] = emb[(size_t)idx[t] * DMODEL + d];
  }
}

__global__ __launch_bounds__(256)
void rmsnorm_kernel(const float* __restrict__ x, const float* __restrict__ w,
                    float* __restrict__ out) {
  __shared__ float red[256];
  const int t = blockIdx.x;
  const float* row = x + (size_t)t * DMODEL;
  float ss = 0.f;
  for (int d = threadIdx.x; d < DMODEL; d += 256) { float a = row[d]; ss += a * a; }
  red[threadIdx.x] = ss;
  __syncthreads();
  for (int off = 128; off > 0; off >>= 1) {
    if (threadIdx.x < off) red[threadIdx.x] += red[threadIdx.x + off];
    __syncthreads();
  }
  const float inv = rsqrtf(red[0] / (float)DMODEL + 1e-6f);
  for (int d = threadIdx.x; d < DMODEL; d += 256)
    out[(size_t)t * DMODEL + d] = row[d] * inv * w[d];
}

__global__ void rope_kernel(float* __restrict__ q, float* __restrict__ kk) {
  const int t = blockIdx.x;
  for (int i = threadIdx.x; i < (NHEAD + NKV) * 32; i += blockDim.x) {
    float* base; int j;
    if (i < NHEAD * 32) {
      base = q + (size_t)t * NHEAD * HEADD + (i >> 5) * HEADD; j = i & 31;
    } else {
      const int ii = i - NHEAD * 32;
      base = kk + (size_t)t * NKV * HEADD + (ii >> 5) * HEADD; j = ii & 31;
    }
    const float inv = powf(10000.0f, -((float)j) / 32.0f);
    const float ang = (float)t * inv;
    const float c = cosf(ang), s = sinf(ang);
    const float x1 = base[j], x2 = base[j + 32];
    base[j]      = x1 * c - x2 * s;
    base[j + 32] = x2 * c + x1 * s;
  }
}

__global__ void gelumul_kernel(float* __restrict__ g, const float* __restrict__ u, int n) {
  for (int i = blockIdx.x * blockDim.x + threadIdx.x; i < n;
       i += gridDim.x * blockDim.x) {
    const float x = g[i];
    const float t = tanhf(0.7978845608028654f * (x + 0.044715f * x * x * x));
    g[i] = 0.5f * x * (1.f + t) * u[i];
  }
}

// ---------------------------------------------------------------- launcher

extern "C" void kernel_launch(void* const* d_in, const int* in_sizes, int n_in,
                              void* d_out, int out_size, void* d_ws, size_t ws_size,
                              hipStream_t stream) {
  (void)in_sizes; (void)n_in; (void)out_size; (void)ws_size;
  const int*   idx   = (const int*)d_in[0];
  const float* embed = (const float*)d_in[1];
  const float* ln1   = (const float*)d_in[2];
  const float* Wq    = (const float*)d_in[3];
  const float* Wk    = (const float*)d_in[4];
  const float* Wv    = (const float*)d_in[5];
  const float* Wo    = (const float*)d_in[6];
  const float* ln2   = (const float*)d_in[7];
  const float* Wg    = (const float*)d_in[8];
  const float* Wu    = (const float*)d_in[9];
  const float* Wd    = (const float*)d_in[10];
  const float* normf = (const float*)d_in[11];
  float* out = (float*)d_out;

  // workspace carve (fp32)
  float* x  = (float*)d_ws;
  float* hb = x  + (size_t)T_SEQ * DMODEL;
  float* qb = hb + (size_t)T_SEQ * DMODEL;
  float* kb = qb + (size_t)T_SEQ * NHEAD * HEADD;
  float* vb = kb + (size_t)T_SEQ * NKV * HEADD;
  float* ab = vb + (size_t)T_SEQ * NKV * HEADD;
  float* gb = ab + (size_t)T_SEQ * NHEAD * HEADD;
  float* ub = gb + (size_t)T_SEQ * FFDIM;

  const dim3 blk(256);
  embed_kernel<<<dim3(512), blk, 0, stream>>>(idx, embed, x, T_SEQ * DMODEL);

  for (int l = 0; l < NLAYER; ++l) {
    rmsnorm_kernel<<<T_SEQ, blk, 0, stream>>>(x, ln1 + l * DMODEL, hb);

    gemm_bf16_wmma<0><<<dim3((NHEAD * HEADD) / BN, T_SEQ / BM), blk, 0, stream>>>(
        hb, Wq + (size_t)l * DMODEL * NHEAD * HEADD, nullptr, qb,
        T_SEQ, NHEAD * HEADD, DMODEL, DMODEL, NHEAD * HEADD, NHEAD * HEADD);
    gemm_bf16_wmma<0><<<dim3((NKV * HEADD) / BN, T_SEQ / BM), blk, 0, stream>>>(
        hb, Wk + (size_t)l * DMODEL * NKV * HEADD, nullptr, kb,
        T_SEQ, NKV * HEADD, DMODEL, DMODEL, NKV * HEADD, NKV * HEADD);
    gemm_bf16_wmma<0><<<dim3((NKV * HEADD) / BN, T_SEQ / BM), blk, 0, stream>>>(
        hb, Wv + (size_t)l * DMODEL * NKV * HEADD, nullptr, vb,
        T_SEQ, NKV * HEADD, DMODEL, DMODEL, NKV * HEADD, NKV * HEADD);

    rope_kernel<<<T_SEQ, blk, 0, stream>>>(qb, kb);
    attn_flash<<<dim3(T_SEQ / 64, NHEAD), dim3(128), 0, stream>>>(qb, kb, vb, ab);

    // x = x + attn @ Wo   (residual fused in epilogue)
    gemm_bf16_wmma<1><<<dim3(DMODEL / BN, T_SEQ / BM), blk, 0, stream>>>(
        ab, Wo + (size_t)l * NHEAD * HEADD * DMODEL, x, x,
        T_SEQ, DMODEL, NHEAD * HEADD, NHEAD * HEADD, DMODEL, DMODEL);

    rmsnorm_kernel<<<T_SEQ, blk, 0, stream>>>(x, ln2 + l * DMODEL, hb);
    gemm_bf16_wmma<0><<<dim3(FFDIM / BN, T_SEQ / BM), blk, 0, stream>>>(
        hb, Wg + (size_t)l * DMODEL * FFDIM, nullptr, gb,
        T_SEQ, FFDIM, DMODEL, DMODEL, FFDIM, FFDIM);
    gemm_bf16_wmma<0><<<dim3(FFDIM / BN, T_SEQ / BM), blk, 0, stream>>>(
        hb, Wu + (size_t)l * DMODEL * FFDIM, nullptr, ub,
        T_SEQ, FFDIM, DMODEL, DMODEL, FFDIM, FFDIM);
    gelumul_kernel<<<dim3(1024), blk, 0, stream>>>(gb, ub, T_SEQ * FFDIM);
    // x = x + (gelu(g)*u) @ Wd
    gemm_bf16_wmma<1><<<dim3(DMODEL / BN, T_SEQ / BM), blk, 0, stream>>>(
        gb, Wd + (size_t)l * FFDIM * DMODEL, x, x,
        T_SEQ, DMODEL, FFDIM, FFDIM, DMODEL, DMODEL);
  }

  rmsnorm_kernel<<<T_SEQ, blk, 0, stream>>>(x, normf, hb);
  // logits = h @ embed.T via TDM-staged tiles (embed is V x D row-major)
  gemm_bt_tdm<<<dim3(NVOCAB / BN, T_SEQ / BM), blk, 0, stream>>>(
      hb, embed, out, T_SEQ, NVOCAB, DMODEL, DMODEL, DMODEL, NVOCAB);
}
